// GATOTFSModule_62216896249975
// MI455X (gfx1250) — matrix operand order repaired and checked by reading, not compiled, not run on hardware
//
#include <hip/hip_runtime.h>
#include <hip/hip_bf16.h>
#include <math.h>

#define B_   8
#define N_   512
#define NODES (B_ * N_)

typedef __attribute__((ext_vector_type(16))) _Float16 v16h;
typedef __attribute__((ext_vector_type(8)))  _Float16 v8h;
typedef __attribute__((ext_vector_type(8)))  float    v8f;
typedef __attribute__((ext_vector_type(4)))  float    v4f;

__device__ __forceinline__ v8f wmma16(v16h a, v16h b, v8f c) {
  return __builtin_amdgcn_wmma_f32_16x16x32_f16(false, a, false, b, (short)0, c,
                                                false, false);
}

// A-matrix (16x32 f16) per-lane K index: lane-half h, element idx 0..15.
// ISA 7.12.2: lanes 0-15: V0..3 -> K0..7, V4..7 -> K16..23; lanes16-31: +8 / +8.
__device__ __forceinline__ int kmapA(int idx, int h) {
  return idx < 8 ? (8 * h + idx) : (16 + 8 * h + (idx - 8));
}

// ---------------------------------------------------------------------------
// Pack W2 (64x32) and W3 (32x8, zero-padded to 32x16) into WMMA B-fragments.
// B-matrix 32x16 f16 layout: lane-half h holds K=8h..8h+7, 16+8h..16+8h+7 via kmapA (2 per VGPR),
// column N = lane%16.
// ---------------------------------------------------------------------------
__global__ void pack_weights(const float* __restrict__ W2,
                             const float* __restrict__ W3,
                             _Float16* __restrict__ w2frag,   // [ks*2+nh][32][16]
                             _Float16* __restrict__ w3frag) { // [32][16]
  int lane = threadIdx.x;          // 0..31
  int h = lane >> 4, n = lane & 15;
  for (int ks = 0; ks < 2; ++ks)
    for (int nh = 0; nh < 2; ++nh) {
      _Float16* dst = w2frag + ((ks * 2 + nh) * 32 + lane) * 16;
#pragma unroll
      for (int idx = 0; idx < 16; ++idx) {
        int k = ks * 32 + kmapA(idx, h);
        int col = nh * 16 + n;
        dst[idx] = (_Float16)W2[k * 32 + col];
      }
    }
  _Float16* d3 = w3frag + lane * 16;
#pragma unroll
  for (int idx = 0; idx < 16; ++idx) {
    int k = kmapA(idx, h);
    d3[idx] = (n < 8) ? (_Float16)W3[k * 8 + n] : (_Float16)0.f;
  }
}

// ---------------------------------------------------------------------------
// Per-node precompute: attention scores, layer-1 halves, GRU h-path.
// ---------------------------------------------------------------------------
__global__ void node_pre(const float* __restrict__ u,
                         const float* __restrict__ W_attn,
                         const float* __restrict__ a_left,
                         const float* __restrict__ a_right,
                         const float* __restrict__ msg_W1,
                         const float* __restrict__ msg_b1,
                         const float* __restrict__ sigma2,
                         const float* __restrict__ gru_Whh,
                         const float* __restrict__ gru_bhh,
                         float* __restrict__ score_i,
                         float* __restrict__ score_j,
                         float* __restrict__ Apre,
                         float* __restrict__ Bpre,
                         float* __restrict__ gh) {
  int node = blockIdx.x * blockDim.x + threadIdx.x;
  if (node >= NODES) return;
  int b = node / N_;
  float uf[8];
#pragma unroll
  for (int f = 0; f < 8; ++f) uf[f] = u[node * 8 + f];

  float si = 0.f, sj = 0.f;
#pragma unroll
  for (int p = 0; p < 16; ++p) {
    float wu = 0.f;
#pragma unroll
    for (int f = 0; f < 8; ++f) wu += uf[f] * W_attn[f * 16 + p];
    si += wu * a_left[p];
    sj += wu * a_right[p];
  }
  score_i[node] = si;
  score_j[node] = sj;

  float sg = sigma2[b];
  for (int k = 0; k < 64; ++k) {
    float a = msg_b1[k] + sg * msg_W1[17 * 64 + k];
    float bb = 0.f;
#pragma unroll
    for (int f = 0; f < 8; ++f) {
      a  += uf[f] * msg_W1[f * 64 + k];
      bb += uf[f] * msg_W1[(8 + f) * 64 + k];
    }
    Apre[(size_t)node * 64 + k] = a;
    Bpre[(size_t)node * 64 + k] = bb;
  }
#pragma unroll
  for (int c = 0; c < 24; ++c) {
    float g = gru_bhh[c];
#pragma unroll
    for (int f = 0; f < 8; ++f) g += uf[f] * gru_Whh[f * 24 + c];
    gh[(size_t)node * 24 + c] = g;
  }
}

// ---------------------------------------------------------------------------
// Heavy kernel: one block per (b,i) row. Softmax over j, then WMMA message
// MLP over 16-edge tiles, aggregate, GRU update.
// ---------------------------------------------------------------------------
__global__ __launch_bounds__(256, 1)
void edge_iter(const float* __restrict__ u_src, float* __restrict__ u_dst,
               const int* __restrict__ adj, const float* __restrict__ edge_feat,
               const float* __restrict__ score_i, const float* __restrict__ score_j,
               const float* __restrict__ Apre, const float* __restrict__ Bpre,
               const float* __restrict__ gh, const float* __restrict__ msg_W1,
               const _Float16* __restrict__ w2frag, const _Float16* __restrict__ w3frag,
               const float* __restrict__ msg_b2, const float* __restrict__ msg_b3,
               const float* __restrict__ gru_Wih, const float* __restrict__ gru_bih) {
  __shared__ __align__(16) float alpha_s[N_];
  __shared__ float red[256];
  __shared__ __align__(16) _Float16 h2buf[8][16 * 32];
  __shared__ float agg_s[8];

  const int t = threadIdx.x;
  const int node = blockIdx.x;      // b*512 + i
  const int b = node >> 9;

  // ---------------- softmax phase ----------------
  const float si = score_i[node];
  const int* adjrow = adj + (size_t)node * N_;
  const float* srj = score_j + (size_t)b * N_;
  float sv[2];
  int mk[2];
#pragma unroll
  for (int q = 0; q < 2; ++q) {
    int j = t + q * 256;
    int m = adjrow[j];
    float beta = si + srj[j];
    beta = beta > 0.f ? beta : 0.2f * beta;
    sv[q] = m ? beta : -1e9f;
    mk[q] = m;
  }
  red[t] = fmaxf(sv[0], sv[1]);
  __syncthreads();
  for (int off = 128; off > 0; off >>= 1) {
    if (t < off) red[t] = fmaxf(red[t], red[t + off]);
    __syncthreads();
  }
  float rowmax = red[0];
  __syncthreads();
  float le = 0.f;
#pragma unroll
  for (int q = 0; q < 2; ++q) {
    float e = mk[q] ? expf(sv[q] - rowmax) : 0.f;
    alpha_s[t + q * 256] = e;
    le += e;
  }
  red[t] = le;
  __syncthreads();
  for (int off = 128; off > 0; off >>= 1) {
    if (t < off) red[t] += red[t + off];
    __syncthreads();
  }
  float inv = red[0] > 0.f ? 1.f / red[0] : 0.f;
  __syncthreads();
#pragma unroll
  for (int q = 0; q < 2; ++q) alpha_s[t + q * 256] *= inv;
  if (t < 8) agg_s[t] = 0.f;
  __syncthreads();

  // ---------------- message phase ----------------
  const int lane = t & 31;
  const int wave = t >> 5;
  const int h = (lane >> 4) & 1;    // lane half
  const int m = lane & 15;          // edge-in-tile (A layout) / column N (B,C layouts)

  // Per-lane constants for the A-fragment build (fixed i for whole block)
  const float* aprow = Apre + (size_t)node * 64;
  const float* werow = msg_W1 + 16 * 64;   // W1 row for edge_feat channel
  float afc[32], wec[32];
#pragma unroll
  for (int ksb = 0; ksb < 2; ++ksb)
#pragma unroll
    for (int idx = 0; idx < 16; ++idx) {
      int k = ksb * 32 + kmapA(idx, h);
      afc[ksb * 16 + idx] = aprow[k];
      wec[ksb * 16 + idx] = werow[k];
    }

  v16h w2f[4];
#pragma unroll
  for (int q = 0; q < 4; ++q)
    w2f[q] = *(const v16h*)(w2frag + (q * 32 + lane) * 16);
  v16h w3f = *(const v16h*)(w3frag + lane * 16);

  v8f c2init[2];
#pragma unroll
  for (int nh = 0; nh < 2; ++nh) {
    float bv = msg_b2[nh * 16 + m];
#pragma unroll
    for (int r = 0; r < 8; ++r) c2init[nh][r] = bv;
  }
  v8f c3init;
  {
    float bv = (m < 8) ? msg_b3[m] : 0.f;
#pragma unroll
    for (int r = 0; r < 8; ++r) c3init[r] = bv;
  }

  const float* bprows = Bpre + (size_t)(b * N_) * 64;
  const float* efrow = edge_feat + (size_t)node * N_;
  _Float16* myh2 = &h2buf[wave][0];
  float acc = 0.f;

  for (int tile = wave; tile < N_ / 16; tile += 8) {
    int j = tile * 16 + m;
    const float* bp = bprows + (size_t)j * 64;
    float e = efrow[j];

    // layer 1: build two A-fragments (K=0..31, 32..63) with fused decomposition
    v16h af[2];
#pragma unroll
    for (int ksb = 0; ksb < 2; ++ksb) {
      v4f b0 = *(const v4f*)(bp + ksb * 32 + 8 * h);
      v4f b1 = *(const v4f*)(bp + ksb * 32 + 8 * h + 4);
      v4f b2v = *(const v4f*)(bp + ksb * 32 + 16 + 8 * h);
      v4f b3v = *(const v4f*)(bp + ksb * 32 + 16 + 8 * h + 4);
      float bl[16];
#pragma unroll
      for (int q = 0; q < 4; ++q) {
        bl[q] = b0[q]; bl[4 + q] = b1[q]; bl[8 + q] = b2v[q]; bl[12 + q] = b3v[q];
      }
#pragma unroll
      for (int idx = 0; idx < 16; ++idx) {
        float v = afc[ksb * 16 + idx] + bl[idx] + e * wec[ksb * 16 + idx];
        v = v > 0.f ? v : 0.f;
        af[ksb][idx] = (_Float16)v;
      }
    }

    // layer 2: (16 edges x 64) @ (64 x 32) via 4 WMMAs, ReLU, stage to LDS
#pragma unroll
    for (int nh = 0; nh < 2; ++nh) {
      v8f c = c2init[nh];
      c = wmma16(af[0], w2f[0 * 2 + nh], c);
      c = wmma16(af[1], w2f[1 * 2 + nh], c);
#pragma unroll
      for (int r = 0; r < 8; ++r) {
        float v = c[r] > 0.f ? c[r] : 0.f;
        myh2[(r + 8 * h) * 32 + nh * 16 + m] = (_Float16)v;  // [edge][chan]
      }
    }
    asm volatile("s_wait_dscnt 0" ::: "memory");  // wave-local C->A relayout

    // layer 3: reload h2 as A-fragment (K=32), 1 WMMA vs padded W3
    v8h p0 = *(const v8h*)(myh2 + m * 32 + 8 * h);
    v8h p1 = *(const v8h*)(myh2 + m * 32 + 16 + 8 * h);
    v16h a3;
#pragma unroll
    for (int q = 0; q < 8; ++q) { a3[q] = p0[q]; a3[8 + q] = p1[q]; }
    v8f out = wmma16(a3, w3f, c3init);

    // accumulate alpha-weighted messages (lane owns feature f=m, edges r+8h)
    v4f al0 = *(const v4f*)(&alpha_s[tile * 16 + 8 * h]);
    v4f al1 = *(const v4f*)(&alpha_s[tile * 16 + 8 * h + 4]);
#pragma unroll
    for (int r = 0; r < 4; ++r) {
      acc += al0[r] * out[r];
      acc += al1[r] * out[4 + r];
    }
  }
  if (m < 8) atomicAdd(&agg_s[m], acc);
  __syncthreads();

  // ---------------- GRU update ----------------
  if (t < 8) {
    int f = t;
    float aggv[8];
#pragma unroll
    for (int k = 0; k < 8; ++k) aggv[k] = agg_s[k];
    float gr = gru_bih[f], gz = gru_bih[8 + f], gn = gru_bih[16 + f];
#pragma unroll
    for (int k = 0; k < 8; ++k) {
      float a = aggv[k];
      gr += a * gru_Wih[k * 24 + f];
      gz += a * gru_Wih[k * 24 + 8 + f];
      gn += a * gru_Wih[k * 24 + 16 + f];
    }
    const float* ghr = gh + (size_t)node * 24;
    float r = 1.f / (1.f + expf(-(gr + ghr[f])));
    float z = 1.f / (1.f + expf(-(gz + ghr[8 + f])));
    float nn = tanhf(gn + r * ghr[16 + f]);
    float uc = u_src[(size_t)node * 8 + f];
    u_dst[(size_t)node * 8 + f] = (1.f - z) * nn + z * uc;
  }
}

// ---------------------------------------------------------------------------
// Readout MLP 8 -> 64 -> 32 -> 2 (trivial FLOPs; scalar per node)
// ---------------------------------------------------------------------------
__global__ void readout(const float* __restrict__ u,
                        const float* __restrict__ W1, const float* __restrict__ b1,
                        const float* __restrict__ W2, const float* __restrict__ b2,
                        const float* __restrict__ W3, const float* __restrict__ b3,
                        float* __restrict__ out) {
  int node = blockIdx.x * blockDim.x + threadIdx.x;
  if (node >= NODES) return;
  float uf[8];
#pragma unroll
  for (int f = 0; f < 8; ++f) uf[f] = u[(size_t)node * 8 + f];
  float h1[64];
  for (int k = 0; k < 64; ++k) {
    float a = b1[k];
#pragma unroll
    for (int f = 0; f < 8; ++f) a += uf[f] * W1[f * 64 + k];
    h1[k] = fmaxf(a, 0.f);
  }
  float h2[32];
  for (int c = 0; c < 32; ++c) {
    float a = b2[c];
    for (int k = 0; k < 64; ++k) a += h1[k] * W2[k * 32 + c];
    h2[c] = fmaxf(a, 0.f);
  }
#pragma unroll
  for (int s = 0; s < 2; ++s) {
    float a = b3[s];
    for (int c = 0; c < 32; ++c) a += h2[c] * W3[c * 2 + s];
    out[(size_t)node * 2 + s] = a;
  }
}

extern "C" void kernel_launch(void* const* d_in, const int* in_sizes, int n_in,
                              void* d_out, int out_size, void* d_ws, size_t ws_size,
                              hipStream_t stream) {
  const float* u0   = (const float*)d_in[0];
  const int*   adj  = (const int*)d_in[1];
  const float* ef   = (const float*)d_in[2];
  const float* sig2 = (const float*)d_in[3];
  const float* Wat  = (const float*)d_in[4];
  const float* al   = (const float*)d_in[5];
  const float* ar   = (const float*)d_in[6];
  const float* mW1  = (const float*)d_in[7];
  const float* mb1  = (const float*)d_in[8];
  const float* mW2  = (const float*)d_in[9];
  const float* mb2  = (const float*)d_in[10];
  const float* mW3  = (const float*)d_in[11];
  const float* mb3  = (const float*)d_in[12];
  const float* gWih = (const float*)d_in[13];
  const float* gWhh = (const float*)d_in[14];
  const float* gbih = (const float*)d_in[15];
  const float* gbhh = (const float*)d_in[16];
  const float* rW1  = (const float*)d_in[17];
  const float* rb1  = (const float*)d_in[18];
  const float* rW2  = (const float*)d_in[19];
  const float* rb2  = (const float*)d_in[20];
  const float* rW3  = (const float*)d_in[21];
  const float* rb3  = (const float*)d_in[22];

  char* ws = (char*)d_ws;
  _Float16* w2frag = (_Float16*)(ws + 0);            //  4096 B
  _Float16* w3frag = (_Float16*)(ws + 4096);         //  1024 B
  float* ubuf0   = (float*)(ws + 8192);              // 131072 B
  float* ubuf1   = (float*)(ws + 8192 + 131072);     // 131072 B
  float* sci     = (float*)(ws + 270336);            //  16384 B
  float* scj     = (float*)(ws + 286720);            //  16384 B
  float* Apre    = (float*)(ws + 303104);            //   1 MB
  float* Bpre    = (float*)(ws + 303104 + 1048576);  //   1 MB
  float* ghb     = (float*)(ws + 303104 + 2097152);  // 393216 B

  pack_weights<<<1, 32, 0, stream>>>(mW2, mW3, w2frag, w3frag);

  const float* usrc = u0;
  for (int it = 0; it < 10; ++it) {
    float* udst = (it & 1) ? ubuf0 : ubuf1;
    node_pre<<<NODES / 256, 256, 0, stream>>>(usrc, Wat, al, ar, mW1, mb1, sig2,
                                              gWhh, gbhh, sci, scj, Apre, Bpre, ghb);
    edge_iter<<<NODES, 256, 0, stream>>>(usrc, udst, adj, ef, sci, scj, Apre,
                                         Bpre, ghb, mW1, w2frag, w3frag, mb2,
                                         mb3, gWih, gbih);
    usrc = udst;
  }
  readout<<<NODES / 256, 256, 0, stream>>>(usrc, rW1, rb1, rW2, rb2, rW3, rb3,
                                           (float*)d_out);
}
